// EnrichedNodeHead_74088185856593
// MI455X (gfx1250) — matrix operand order repaired and checked by reading, not compiled, not run on hardware
//
#include <hip/hip_runtime.h>
#include <hip/hip_bf16.h>

#define D   64
#define NCI 10

typedef __attribute__((ext_vector_type(16))) __bf16 v16bf;
typedef __attribute__((ext_vector_type(8)))  __bf16 v8bf;
typedef __attribute__((ext_vector_type(8)))  float  v8f;

// ---- packed-weight workspace offsets (in bf16 elements) ----
// layout per weight: tile(tk*NT+tn) -> 32 lanes -> 16 halves
//   lane n   = tn*16 + (lane&15)
//   lane k   = tk*32 + (lane>>4)*8 + i (i<8), tk*32 + 16 + (lane>>4)*8 + (i-8) (i>=8)
static constexpr size_t OFF_WIN  = 0;      // (192,64)  NT=12 KT=2  -> 12288
static constexpr size_t OFF_WOUT = 12288;  // (64,64)   NT=4  KT=2  ->  4096
static constexpr size_t OFF_WI1  = 16384;  // (128,384) NT=8  KT=12 -> 49152
static constexpr size_t OFF_WI2  = 65536;  // (64,128)  NT=4  KT=4  ->  8192
static constexpr size_t OFF_WC1  = 73728;  // (64,10)   NT=4  KT=1  ->  2048 (K zero-padded)
static constexpr size_t OFF_WC2  = 75776;  // (64,64)   NT=4  KT=2  ->  4096
static constexpr size_t OFF_WM   = 79872;  // (64,192)  NT=4  KT=6  -> 12288
static constexpr size_t OFF_WK1  = 92160;  // (64,64)   NT=4  KT=2  ->  4096
static constexpr size_t OFF_WK2  = 96256;  // (8,64)    NT=1  KT=2  ->  1024 (N zero-padded)

static __device__ inline float  bf2f(__bf16 x) { return (float)x; }
static __device__ inline __bf16 f2bf(float  x) { return (__bf16)x; }

static __device__ inline void wave_fence() {
  __builtin_amdgcn_wave_barrier();
  asm volatile("" ::: "memory");
}

static __device__ inline float gelu_erf(float x) {
  return 0.5f * x * (1.0f + erff(x * 0.70710678118654752f));
}

static __device__ inline v8f zero8() {
  v8f z = {0.f, 0.f, 0.f, 0.f, 0.f, 0.f, 0.f, 0.f};
  return z;
}

static __device__ inline v8f wmma_bf16(v16bf a, v16bf b, v8f c) {
  return __builtin_amdgcn_wmma_f32_16x16x32_bf16(false, a, false, b, (short)0, c,
                                                 false, false);
}

// A fragment (16x32 bf16) from LDS row-major [16][S] bf16, K offset k0.
// Layout: lanes 0-15 M=0..15 hold K {k0..k0+7, k0+16..k0+23},
//         lanes 16-31 same M   hold K {k0+8..k0+15, k0+24..k0+31}.
static __device__ inline v16bf afrag_lds(const __bf16* A, int S, int lane, int k0) {
  int row = lane & 15, hi = lane >> 4;
  const v8bf* p0 = (const v8bf*)(A + row * S + k0 + hi * 8);
  const v8bf* p1 = (const v8bf*)(A + row * S + k0 + 16 + hi * 8);
  v8bf a = *p0, b = *p1;
  return __builtin_shufflevector(a, b, 0, 1, 2, 3, 4, 5, 6, 7,
                                 8, 9, 10, 11, 12, 13, 14, 15);
}

// A fragment built from global fp32 rows (row stride D=64), converted to bf16.
static __device__ inline v16bf afrag_global_f32(const float* E, int m0, int lane, int k0) {
  int row = lane & 15, hi = lane >> 4;
  const float4* q0 = (const float4*)(E + (size_t)(m0 + row) * D + k0 + hi * 8);
  const float4* q1 = (const float4*)(E + (size_t)(m0 + row) * D + k0 + 16 + hi * 8);
  float4 f0 = q0[0], f1 = q0[1], f2 = q1[0], f3 = q1[1];
  v16bf r;
  r[0]  = f2bf(f0.x); r[1]  = f2bf(f0.y); r[2]  = f2bf(f0.z); r[3]  = f2bf(f0.w);
  r[4]  = f2bf(f1.x); r[5]  = f2bf(f1.y); r[6]  = f2bf(f1.z); r[7]  = f2bf(f1.w);
  r[8]  = f2bf(f2.x); r[9]  = f2bf(f2.y); r[10] = f2bf(f2.z); r[11] = f2bf(f2.w);
  r[12] = f2bf(f3.x); r[13] = f2bf(f3.y); r[14] = f2bf(f3.z); r[15] = f2bf(f3.w);
  return r;
}

// B fragment (32x16 bf16) from packed weights: 16 contiguous halves per lane.
static __device__ inline v16bf bfrag(const __bf16* Wp, int NT, int tn, int tk, int lane) {
  const v8bf* p = (const v8bf*)(Wp + (((size_t)(tk * NT + tn)) * 32 + lane) * 16);
  v8bf a = p[0], b = p[1];
  return __builtin_shufflevector(a, b, 0, 1, 2, 3, 4, 5, 6, 7,
                                 8, 9, 10, 11, 12, 13, 14, 15);
}

// -------------------- weight pack kernel --------------------
__global__ void pack_weight_kernel(const float* __restrict__ W, __bf16* __restrict__ dst,
                                   int Nout, int Kin, int NT, int KT) {
  int idx = blockIdx.x * blockDim.x + threadIdx.x;
  int tot = NT * KT * 512;
  if (idx >= tot) return;
  int i    = idx & 15;
  int lane = (idx >> 4) & 31;
  int tile = idx >> 9;
  int tn = tile % NT, tk = tile / NT;
  int n = tn * 16 + (lane & 15);
  int k = tk * 32 + (lane >> 4) * 8 + (i < 8 ? i : 8 + i);
  float v = (n < Nout && k < Kin) ? W[(size_t)n * Kin + k] : 0.f;
  dst[idx] = f2bf(v);
}

// -------------------- LN helper (16 rows x 64 cols) --------------------
static __device__ inline void ln_rows16(const float* s_ln, const float* g, const float* b,
                                        __bf16* dst, int dstS, int dstOff, int lane,
                                        bool do_gelu) {
  if (lane < 16) {
    const float* x = s_ln + lane * 64;
    float s = 0.f, ss = 0.f;
    for (int c = 0; c < 64; ++c) { float v = x[c]; s += v; ss += v * v; }
    float mu  = s * (1.f / 64.f);
    float var = ss * (1.f / 64.f) - mu * mu;
    float rs  = rsqrtf(var + 1e-5f);
    for (int c = 0; c < 64; ++c) {
      float y = (x[c] - mu) * rs * g[c] + b[c];
      if (do_gelu) y = gelu_erf(y);
      dst[lane * dstS + dstOff + c] = f2bf(y);
    }
  }
}

// -------------------- main fused kernel: 1 wave32 = 16 nodes --------------------
__global__ __launch_bounds__(32)
void enriched_node_head_kernel(
    const float* __restrict__ evx, const float* __restrict__ evy,
    const float* __restrict__ exv, const float* __restrict__ eyv,
    const float* __restrict__ ci,
    const float* __restrict__ b_in, const float* __restrict__ b_out,
    const float* __restrict__ g_attn, const float* __restrict__ b_attn,
    const float* __restrict__ bi1, const float* __restrict__ bi2,
    const float* __restrict__ gi,  const float* __restrict__ bni,
    const float* __restrict__ bc1, const float* __restrict__ bc2,
    const float* __restrict__ gc,  const float* __restrict__ bnc,
    const float* __restrict__ bm,  const float* __restrict__ gm,
    const float* __restrict__ bnm,
    const float* __restrict__ bk1, const float* __restrict__ bk2,
    const __bf16* __restrict__ Wp,
    float* __restrict__ out) {
  const int lane = threadIdx.x;
  const int m0   = blockIdx.x * 16;
  const int nl   = lane & 15;
  const int hi   = lane >> 4;

  // 55808 bytes of per-wave LDS, regions overlaid across phases.
  __shared__ __align__(16) unsigned char smem[55808];
  __bf16* s_qkv  = (__bf16*)(smem);          // [4][16][192] bf16 (phase A)
  __bf16* s_ao   = (__bf16*)(smem + 24576);  // [4][16][64]  bf16 (phase A)
  float*  s_attf = (float*)(smem + 32768);   // [4][16][64]  f32  (phase A)
  float*  s_stat = (float*)(smem + 49152);   // [4][16][2]   f32  mu/rstd
  __bf16* s_act  = (__bf16*)(smem + 49664);  // [16][192]    bf16 merge input
  __bf16* s_bufA = (__bf16*)(smem);          // overlay: up to [16][384] bf16
  __bf16* s_bufB = (__bf16*)(smem + 24576);  // overlay: up to [16][128] bf16
  float*  s_ln   = (float*)(smem + 32768);   // overlay: [16][64] f32

  const float* E[4] = {evx, evy, exv, eyv};

  // ---------------- Phase A: QKV projection (per-token GEMMs) ----------------
#pragma unroll
  for (int t = 0; t < 4; ++t) {
    v16bf a0 = afrag_global_f32(E[t], m0, lane, 0);
    v16bf a1 = afrag_global_f32(E[t], m0, lane, 32);
#pragma unroll 1
    for (int j = 0; j < 12; ++j) {
      v8f acc = zero8();
      acc = wmma_bf16(a0, bfrag(Wp + OFF_WIN, 12, j, 0, lane), acc);
      acc = wmma_bf16(a1, bfrag(Wp + OFF_WIN, 12, j, 1, lane), acc);
      float bias = b_in[j * 16 + nl];
#pragma unroll
      for (int r = 0; r < 8; ++r) {
        int row = r + 8 * hi;
        s_qkv[(t * 16 + row) * 192 + j * 16 + nl] = f2bf(acc[r] + bias);
      }
    }
  }
  wave_fence();

  // ------------- attention over 4 tokens (fp32 VALU, 8 tasks/lane) -------------
#pragma unroll 1
  for (int i = 0; i < 8; ++i) {
    int task = lane + 32 * i;          // 0..255 : (node, head, query-token)
    int m  = task & 15;
    int h  = (task >> 4) & 3;
    int tq = task >> 6;
    const __bf16* qp = s_qkv + (tq * 16 + m) * 192 + h * 16;
    float q[16];
#pragma unroll
    for (int e = 0; e < 16; ++e) q[e] = bf2f(qp[e]);
    float sc[4];
#pragma unroll
    for (int tk = 0; tk < 4; ++tk) {
      const __bf16* kp = s_qkv + (tk * 16 + m) * 192 + 64 + h * 16;
      float s = 0.f;
#pragma unroll
      for (int e = 0; e < 16; ++e) s += q[e] * bf2f(kp[e]);
      sc[tk] = s * 0.25f;              // 1/sqrt(hd=16)
    }
    float mx  = fmaxf(fmaxf(sc[0], sc[1]), fmaxf(sc[2], sc[3]));
    float den = 0.f;
#pragma unroll
    for (int tk = 0; tk < 4; ++tk) { sc[tk] = __expf(sc[tk] - mx); den += sc[tk]; }
    float inv = 1.f / den;
    float o[16];
#pragma unroll
    for (int e = 0; e < 16; ++e) o[e] = 0.f;
#pragma unroll
    for (int tk = 0; tk < 4; ++tk) {
      const __bf16* vp = s_qkv + (tk * 16 + m) * 192 + 128 + h * 16;
      float a = sc[tk] * inv;
#pragma unroll
      for (int e = 0; e < 16; ++e) o[e] += a * bf2f(vp[e]);
    }
    __bf16* op = s_ao + (tq * 16 + m) * 64 + h * 16;
#pragma unroll
    for (int e = 0; e < 16; ++e) op[e] = f2bf(o[e]);
  }
  wave_fence();

  // ---------------- out-projection + residual (per token) ----------------
#pragma unroll
  for (int t = 0; t < 4; ++t) {
    v16bf a0 = afrag_lds(s_ao + t * 16 * 64, 64, lane, 0);
    v16bf a1 = afrag_lds(s_ao + t * 16 * 64, 64, lane, 32);
#pragma unroll 1
    for (int j = 0; j < 4; ++j) {
      v8f acc = zero8();
      acc = wmma_bf16(a0, bfrag(Wp + OFF_WOUT, 4, j, 0, lane), acc);
      acc = wmma_bf16(a1, bfrag(Wp + OFF_WOUT, 4, j, 1, lane), acc);
      float bias = b_out[j * 16 + nl];
#pragma unroll
      for (int r = 0; r < 8; ++r) {
        int row  = r + 8 * hi;
        float rd = E[t][(size_t)(m0 + row) * D + j * 16 + nl];
        s_attf[(t * 16 + row) * 64 + j * 16 + nl] = acc[r] + bias + rd;
      }
    }
  }
  wave_fence();

  // ---------------- LN (per token-row) + mean pool -> s_act[:, 0:64] ----------------
#pragma unroll
  for (int i = 0; i < 2; ++i) {
    int task = lane + 32 * i;          // 64 (token,row) pairs
    int m = task & 15, t = task >> 4;
    const float* x = s_attf + (t * 16 + m) * 64;
    float s = 0.f, ss = 0.f;
    for (int c = 0; c < 64; ++c) { float v = x[c]; s += v; ss += v * v; }
    float mu  = s * (1.f / 64.f);
    float var = ss * (1.f / 64.f) - mu * mu;
    s_stat[(t * 16 + m) * 2 + 0] = mu;
    s_stat[(t * 16 + m) * 2 + 1] = rsqrtf(var + 1e-5f);
  }
  wave_fence();
  {
    int m = nl;
    for (int cc = 0; cc < 32; ++cc) {
      int c = hi * 32 + cc;
      float g = g_attn[c], bb = b_attn[c];
      float acc = 0.f;
#pragma unroll
      for (int t = 0; t < 4; ++t) {
        float mu = s_stat[(t * 16 + m) * 2 + 0];
        float rs = s_stat[(t * 16 + m) * 2 + 1];
        acc += (s_attf[(t * 16 + m) * 64 + c] - mu) * rs * g + bb;
      }
      s_act[m * 192 + c] = f2bf(acc * 0.25f);
    }
  }
  wave_fence();

  // ---------------- Phase B: interaction features 16x384 ----------------
  {
    const int pa[6] = {0, 0, 0, 1, 1, 2};
    const int pb[6] = {1, 2, 3, 2, 3, 3};
    int m = nl;
#pragma unroll
    for (int pp = 0; pp < 3; ++pp) {
      int p = hi * 3 + pp;
      const float* A_ = E[pa[p]] + (size_t)(m0 + m) * D;
      const float* B_ = E[pb[p]] + (size_t)(m0 + m) * D;
      for (int e = 0; e < 64; ++e)
        s_bufA[m * 384 + p * 64 + e] = f2bf(A_[e] * B_[e]);
    }
  }
  wave_fence();
#pragma unroll 1
  for (int j = 0; j < 8; ++j) {           // Wi1: 384 -> 128, GELU
    v8f acc = zero8();
#pragma unroll 1
    for (int tk = 0; tk < 12; ++tk) {
      v16bf a = afrag_lds(s_bufA, 384, lane, tk * 32);
      acc = wmma_bf16(a, bfrag(Wp + OFF_WI1, 8, j, tk, lane), acc);
    }
    float bias = bi1[j * 16 + nl];
#pragma unroll
    for (int r = 0; r < 8; ++r)
      s_bufB[(r + 8 * hi) * 128 + j * 16 + nl] = f2bf(gelu_erf(acc[r] + bias));
  }
  wave_fence();
#pragma unroll 1
  for (int j = 0; j < 4; ++j) {           // Wi2: 128 -> 64
    v8f acc = zero8();
#pragma unroll
    for (int tk = 0; tk < 4; ++tk) {
      v16bf a = afrag_lds(s_bufB, 128, lane, tk * 32);
      acc = wmma_bf16(a, bfrag(Wp + OFF_WI2, 4, j, tk, lane), acc);
    }
    float bias = bi2[j * 16 + nl];
#pragma unroll
    for (int r = 0; r < 8; ++r)
      s_ln[(r + 8 * hi) * 64 + j * 16 + nl] = acc[r] + bias;
  }
  wave_fence();
  ln_rows16(s_ln, gi, bni, s_act, 192, 64, lane, false);
  wave_fence();

  // ---------------- Phase C: CI projector (K padded 10 -> 32) ----------------
  if (lane < 16) {
    const float* cp = ci + (size_t)(m0 + lane) * NCI;
    for (int c = 0; c < NCI; ++c) s_bufA[lane * 32 + c] = f2bf(cp[c]);
    for (int c = NCI; c < 32; ++c) s_bufA[lane * 32 + c] = f2bf(0.f);
  }
  wave_fence();
#pragma unroll 1
  for (int j = 0; j < 4; ++j) {           // Wc1: 10 -> 64, GELU
    v16bf a = afrag_lds(s_bufA, 32, lane, 0);
    v8f acc = wmma_bf16(a, bfrag(Wp + OFF_WC1, 4, j, 0, lane), zero8());
    float bias = bc1[j * 16 + nl];
#pragma unroll
    for (int r = 0; r < 8; ++r)
      s_bufB[(r + 8 * hi) * 64 + j * 16 + nl] = f2bf(gelu_erf(acc[r] + bias));
  }
  wave_fence();
#pragma unroll 1
  for (int j = 0; j < 4; ++j) {           // Wc2: 64 -> 64
    v8f acc = zero8();
#pragma unroll
    for (int tk = 0; tk < 2; ++tk) {
      v16bf a = afrag_lds(s_bufB, 64, lane, tk * 32);
      acc = wmma_bf16(a, bfrag(Wp + OFF_WC2, 4, j, tk, lane), acc);
    }
    float bias = bc2[j * 16 + nl];
#pragma unroll
    for (int r = 0; r < 8; ++r)
      s_ln[(r + 8 * hi) * 64 + j * 16 + nl] = acc[r] + bias;
  }
  wave_fence();
  ln_rows16(s_ln, gc, bnc, s_act, 192, 128, lane, false);
  wave_fence();

  // ---------------- Phase D: merge 192 -> 64, LN, GELU ----------------
#pragma unroll 1
  for (int j = 0; j < 4; ++j) {
    v8f acc = zero8();
#pragma unroll 1
    for (int tk = 0; tk < 6; ++tk) {
      v16bf a = afrag_lds(s_act, 192, lane, tk * 32);
      acc = wmma_bf16(a, bfrag(Wp + OFF_WM, 4, j, tk, lane), acc);
    }
    float bias = bm[j * 16 + nl];
#pragma unroll
    for (int r = 0; r < 8; ++r)
      s_ln[(r + 8 * hi) * 64 + j * 16 + nl] = acc[r] + bias;
  }
  wave_fence();
  ln_rows16(s_ln, gm, bnm, s_bufB, 64, 0, lane, true);  // LN then GELU
  wave_fence();

  // ---------------- Phase E: classifier ----------------
#pragma unroll 1
  for (int j = 0; j < 4; ++j) {           // Wk1: 64 -> 64, GELU
    v8f acc = zero8();
#pragma unroll
    for (int tk = 0; tk < 2; ++tk) {
      v16bf a = afrag_lds(s_bufB, 64, lane, tk * 32);
      acc = wmma_bf16(a, bfrag(Wp + OFF_WK1, 4, j, tk, lane), acc);
    }
    float bias = bk1[j * 16 + nl];
#pragma unroll
    for (int r = 0; r < 8; ++r)
      s_bufA[(r + 8 * hi) * 64 + j * 16 + nl] = f2bf(gelu_erf(acc[r] + bias));
  }
  wave_fence();
  {                                       // Wk2: 64 -> 8 (N padded to 16)
    v8f acc = zero8();
    v16bf a0 = afrag_lds(s_bufA, 64, lane, 0);
    v16bf a1 = afrag_lds(s_bufA, 64, lane, 32);
    acc = wmma_bf16(a0, bfrag(Wp + OFF_WK2, 1, 0, 0, lane), acc);
    acc = wmma_bf16(a1, bfrag(Wp + OFF_WK2, 1, 0, 1, lane), acc);
    if (nl < 8) {
      float bias = bk2[nl];
#pragma unroll
      for (int r = 0; r < 8; ++r) {
        int row = r + 8 * hi;
        out[(size_t)(m0 + row) * 8 + nl] = acc[r] + bias;
      }
    }
  }
}

extern "C" void kernel_launch(void* const* d_in, const int* in_sizes, int n_in,
                              void* d_out, int out_size, void* d_ws, size_t ws_size,
                              hipStream_t stream) {
  const float* evx = (const float*)d_in[0];
  const float* evy = (const float*)d_in[1];
  const float* exv = (const float*)d_in[2];
  const float* eyv = (const float*)d_in[3];
  const float* ci  = (const float*)d_in[4];

  __bf16* wp = (__bf16*)d_ws;

  auto pack = [&](int widx, size_t off, int Nout, int Kin, int NT, int KT) {
    int tot = NT * KT * 512;
    pack_weight_kernel<<<(tot + 255) / 256, 256, 0, stream>>>(
        (const float*)d_in[widx], wp + off, Nout, Kin, NT, KT);
  };
  pack(5,  OFF_WIN,  192, 64,  12, 2);   // W_in
  pack(7,  OFF_WOUT, 64,  64,  4,  2);   // W_out
  pack(11, OFF_WI1,  128, 384, 8,  12);  // Wi1
  pack(13, OFF_WI2,  64,  128, 4,  4);   // Wi2
  pack(17, OFF_WC1,  64,  10,  4,  1);   // Wc1 (K zero-padded to 32)
  pack(19, OFF_WC2,  64,  64,  4,  2);   // Wc2
  pack(23, OFF_WM,   64,  192, 4,  6);   // Wm
  pack(27, OFF_WK1,  64,  64,  4,  2);   // Wk1
  pack(29, OFF_WK2,  8,   64,  1,  2);   // Wk2 (N zero-padded to 16)

  int N = in_sizes[0] / D;               // 131072
  enriched_node_head_kernel<<<N / 16, 32, 0, stream>>>(
      evx, evy, exv, eyv, ci,
      (const float*)d_in[6],  (const float*)d_in[8],
      (const float*)d_in[9],  (const float*)d_in[10],
      (const float*)d_in[12], (const float*)d_in[14],
      (const float*)d_in[15], (const float*)d_in[16],
      (const float*)d_in[18], (const float*)d_in[20],
      (const float*)d_in[21], (const float*)d_in[22],
      (const float*)d_in[24], (const float*)d_in[25],
      (const float*)d_in[26],
      (const float*)d_in[28], (const float*)d_in[30],
      wp, (float*)d_out);
}
